// ContrastiveQueueLoss_32916629356880
// MI455X (gfx1250) — matrix operand order repaired and checked by reading, not compiled, not run on hardware
//
#include <hip/hip_runtime.h>
#include <stdint.h>

// ============================================================================
// ContrastiveQueueLoss for MI455X (gfx1250, wave32, WMMA)
//
// V(2,256,128), L(256,128), queue(65536,128) -> scalar f32 loss.
// Dominant work: 512x65536x128 GEMM fused with row-wise sum-of-exp
// (|logit| <= 10 -> plain exp-sum is f32-safe, no max subtraction).
//
// Precision: f32 split into bf16 hi + lo; hi*hi + hi*lo + lo*hi with
// v_wmma_f32_16x16x32_bf16 (f32 accumulate) ~= fp32-grade dot products at
// bf16-WMMA throughput (26 GFLOP total; queue splits live in L2).
//
// Two q-tiles in flight per wave (independent accumulators c0/c1, WMMAs
// alternate between them) so the exp epilogue of one tile overlaps the WMMA
// chain of the other and WMMA->VALU hazard NOPs are hidden;
// exp(10x) folded to exp2(x * 10*log2e) -> single v_mul + v_exp_f32
// (via __builtin_amdgcn_exp2f, the native exp2 VALU op).
//
// Workspace (~33.95 MB): Vhi|Vlo (128KB each), Qhi|Qlo (16MB each),
// pos[512] f32, partial[32*64*16] f32 (deterministic strip partials).
// ============================================================================

typedef __attribute__((ext_vector_type(16))) __bf16 v16bf;
typedef __attribute__((ext_vector_type(8)))  float  v8f;

#define EXP2_SCALE 14.42695040888963f   // 10 / ln(2):  exp(10x) = exp2(x*this)

static __device__ __forceinline__ unsigned short f32_to_bf16_rne(float f) {
    unsigned int u = __float_as_uint(f);
    u += 0x7FFFu + ((u >> 16) & 1u);          // round to nearest even
    return (unsigned short)(u >> 16);
}
static __device__ __forceinline__ float bf16_to_f32(unsigned short h) {
    return __uint_as_float(((unsigned int)h) << 16);
}

// ---------------------------------------------------------------------------
// Split f32 array into bf16 hi/lo arrays (two elements/thread, dword stores).
// ---------------------------------------------------------------------------
__global__ __launch_bounds__(256)
void cvt_split_kernel(const float* __restrict__ src,
                      unsigned int* __restrict__ hi,
                      unsigned int* __restrict__ lo, int npair) {
    int t = blockIdx.x * 256 + threadIdx.x;
    if (t >= npair) return;
    float2 x = ((const float2*)src)[t];
    unsigned short h0 = f32_to_bf16_rne(x.x);
    unsigned short h1 = f32_to_bf16_rne(x.y);
    unsigned short l0 = f32_to_bf16_rne(x.x - bf16_to_f32(h0));
    unsigned short l1 = f32_to_bf16_rne(x.y - bf16_to_f32(h1));
    hi[t] = (unsigned int)h0 | ((unsigned int)h1 << 16);
    lo[t] = (unsigned int)l0 | ((unsigned int)l1 << 16);
}

// ---------------------------------------------------------------------------
// pos[jb] = 10 * dot(V[jb], L[jb%256])   (full f32 precision; tiny)
// ---------------------------------------------------------------------------
__global__ __launch_bounds__(256)
void pos_kernel(const float* __restrict__ V, const float* __restrict__ L,
                float* __restrict__ pos) {
    int t = blockIdx.x * 256 + threadIdx.x;          // 0..511
    const float4* v = (const float4*)(V + (size_t)t * 128);
    const float4* l = (const float4*)(L + (size_t)(t & 255) * 128);
    float s = 0.0f;
#pragma unroll
    for (int k = 0; k < 32; ++k) {
        float4 a = v[k], b = l[k];
        s += a.x * b.x + a.y * b.y + a.z * b.z + a.w * b.w;
    }
    pos[t] = s * 10.0f;
}

// ---------------------------------------------------------------------------
// Fused GEMM + exp-reduction.  Per (row-tile, strip) block, each of 8 waves
// owns 8 q-tiles processed as 4 pairs with independent accumulators.
//
// A-frag (16x32 bf16): lane<16 row M=lane, K=kc..kc+15;
//                      lane>=16 same row, K=kc+16..kc+31. -> two b128 loads.
// B-frag (32x16 bf16): lane<16 col N=lane (queue row), K=kc..kc+15;
//                      lane>=16 K=kc+16..kc+31.     -> two b128 loads.
// C/D (16x16 f32): vgpr i = row i (low lanes) / i+8 (high), col = lane%16.
// ---------------------------------------------------------------------------
__global__ __launch_bounds__(256)
void gemm_lse_kernel(const unsigned short* __restrict__ Vhi,
                     const unsigned short* __restrict__ Vlo,
                     const unsigned short* __restrict__ Qhi,
                     const unsigned short* __restrict__ Qlo,
                     float* __restrict__ partial) {
    const int rt    = blockIdx.x;        // 0..31  row tile (16 rows of V_flat)
    const int strip = blockIdx.y;        // 0..63  strip of 64 q-tiles
    const int wave  = threadIdx.x >> 5;  // 0..7
    const int lane  = threadIdx.x & 31;
    const int half  = lane >> 4;         // 0 or 1
    const int ln16  = lane & 15;

    union Frag { uint4 u[2]; v16bf f; };

    // A fragments for this row tile: resident in VGPRs for the whole kernel.
    Frag ahi[4], alo[4];
    {
        const unsigned short* arh = Vhi + (size_t)(rt * 16 + ln16) * 128;
        const unsigned short* arl = Vlo + (size_t)(rt * 16 + ln16) * 128;
#pragma unroll
        for (int k = 0; k < 4; ++k) {
            const int e0 = k * 32 + half * 16;
            const int e1 = k * 32 + half * 16 + 8;
            ahi[k].u[0] = *(const uint4*)(arh + e0);
            ahi[k].u[1] = *(const uint4*)(arh + e1);
            alo[k].u[0] = *(const uint4*)(arl + e0);
            alo[k].u[1] = *(const uint4*)(arl + e1);
        }
    }

    float acc[8];
#pragma unroll
    for (int i = 0; i < 8; ++i) acc[i] = 0.0f;

    const int qt_base = strip * 64 + wave * 8;       // 8 q-tiles per wave
    const unsigned short* brh0_base =
        Qhi + (size_t)(qt_base * 16 + ln16) * 128 + half * 16;
    const unsigned short* brl0_base =
        Qlo + (size_t)(qt_base * 16 + ln16) * 128 + half * 16;

#pragma unroll 1
    for (int qq = 0; qq < 4; ++qq) {                 // 2 tiles per iteration
        const unsigned short* brh0 = brh0_base + (size_t)qq * 4096;
        const unsigned short* brl0 = brl0_base + (size_t)qq * 4096;
        const unsigned short* brh1 = brh0 + 2048;    // second tile of the pair
        const unsigned short* brl1 = brl0 + 2048;

        if (qq < 3) {                                // hint next pair into L2
            __builtin_prefetch(brh0 + 4096, 0, 1);   // global_prefetch_b8
            __builtin_prefetch(brl0 + 4096, 0, 1);
            __builtin_prefetch(brh1 + 4096, 0, 1);
            __builtin_prefetch(brl1 + 4096, 0, 1);
        }

        v8f c0 = {0.f, 0.f, 0.f, 0.f, 0.f, 0.f, 0.f, 0.f};
        v8f c1 = {0.f, 0.f, 0.f, 0.f, 0.f, 0.f, 0.f, 0.f};
#pragma unroll
        for (int k = 0; k < 4; ++k) {
            Frag bh0, bl0, bh1, bl1;
            bh0.u[0] = *(const uint4*)(brh0 + 32 * k);
            bh0.u[1] = *(const uint4*)(brh0 + 32 * k + 8);
            bh1.u[0] = *(const uint4*)(brh1 + 32 * k);
            bh1.u[1] = *(const uint4*)(brh1 + 32 * k + 8);
            bl0.u[0] = *(const uint4*)(brl0 + 32 * k);
            bl0.u[1] = *(const uint4*)(brl0 + 32 * k + 8);
            bl1.u[0] = *(const uint4*)(brl1 + 32 * k);
            bl1.u[1] = *(const uint4*)(brl1 + 32 * k + 8);
            // Alternate accumulators: no back-to-back RAW on the same D.
            c0 = __builtin_amdgcn_wmma_f32_16x16x32_bf16(
                     false, ahi[k].f, false, bh0.f, (short)0, c0, false, false);
            c1 = __builtin_amdgcn_wmma_f32_16x16x32_bf16(
                     false, ahi[k].f, false, bh1.f, (short)0, c1, false, false);
            c0 = __builtin_amdgcn_wmma_f32_16x16x32_bf16(
                     false, ahi[k].f, false, bl0.f, (short)0, c0, false, false);
            c1 = __builtin_amdgcn_wmma_f32_16x16x32_bf16(
                     false, ahi[k].f, false, bl1.f, (short)0, c1, false, false);
            c0 = __builtin_amdgcn_wmma_f32_16x16x32_bf16(
                     false, alo[k].f, false, bh0.f, (short)0, c0, false, false);
            c1 = __builtin_amdgcn_wmma_f32_16x16x32_bf16(
                     false, alo[k].f, false, bh1.f, (short)0, c1, false, false);
        }
        // Fused epilogue: exp(10x) = exp2(x * 10*log2e); one mul + v_exp_f32.
#pragma unroll
        for (int i = 0; i < 8; ++i)
            acc[i] += __builtin_amdgcn_exp2f(c0[i] * EXP2_SCALE);
#pragma unroll
        for (int i = 0; i < 8; ++i)
            acc[i] += __builtin_amdgcn_exp2f(c1[i] * EXP2_SCALE);
    }

    // Reduce across the 16 columns (lanes within each wave half).
#pragma unroll
    for (int m = 1; m < 16; m <<= 1)
#pragma unroll
        for (int i = 0; i < 8; ++i)
            acc[i] += __shfl_xor(acc[i], m, 32);

    __shared__ float red[8][16];
    if (ln16 == 0) {
#pragma unroll
        for (int i = 0; i < 8; ++i) red[wave][half * 8 + i] = acc[i];
    }
    __syncthreads();
    if (threadIdx.x < 16) {                          // deterministic wave sum
        float s = 0.0f;
#pragma unroll
        for (int w = 0; w < 8; ++w) s += red[w][threadIdx.x];
        partial[(size_t)(rt * 64 + strip) * 16 + threadIdx.x] = s;
    }
}

// ---------------------------------------------------------------------------
// Final: P[j] = sum_b exp(pos[j,b]);  rowQ[jb] = sum_strips partial;
// loss = -(1/256) * sum_jb (pos - log(P[j] + rowQ[jb])).  Deterministic trees.
// ---------------------------------------------------------------------------
__global__ __launch_bounds__(512)
void finalize_kernel(const float* __restrict__ pos,
                     const float* __restrict__ partial,
                     float* __restrict__ out) {
    __shared__ float buf[512];
    const int t = threadIdx.x;           // row index jb, 0..511
    const int b = t & 255;

    float p = pos[t];
    buf[t] = __expf(p);
    __syncthreads();
    for (int s = 128; s > 0; s >>= 1) {  // per-j (per-256-half) reduction
        if (b < s) buf[t] += buf[t + s];
        __syncthreads();
    }
    const float P = buf[(t >> 8) << 8];  // P[j]
    __syncthreads();

    float q = 0.0f;                      // fixed-order strip reduction
    const float* pp = partial + (size_t)(t >> 4) * 1024 + (t & 15);
    for (int strip = 0; strip < 64; ++strip) q += pp[strip * 16];

    buf[t] = p - __logf(P + q);
    __syncthreads();
    for (int s = 256; s > 0; s >>= 1) {
        if (t < s) buf[t] += buf[t + s];
        __syncthreads();
    }
    if (t == 0) out[0] = -buf[0] * (1.0f / 256.0f);
}

// ---------------------------------------------------------------------------
extern "C" void kernel_launch(void* const* d_in, const int* in_sizes, int n_in,
                              void* d_out, int out_size, void* d_ws, size_t ws_size,
                              hipStream_t stream) {
    (void)in_sizes; (void)n_in; (void)out_size;
    const float* V     = (const float*)d_in[0];   // (2,256,128)
    const float* L     = (const float*)d_in[1];   // (256,128)
    const float* queue = (const float*)d_in[2];   // (65536,128)
    float* out = (float*)d_out;

    // Workspace carve-up (bytes)
    const size_t OFF_VHI = 0;
    const size_t OFF_VLO = 131072;                 // 512*128*2
    const size_t OFF_QHI = 262144;
    const size_t OFF_QLO = OFF_QHI + 16777216;     // 65536*128*2
    const size_t OFF_POS = OFF_QLO + 16777216;
    const size_t OFF_PAR = OFF_POS + 512 * 4;
    const size_t WS_NEED = OFF_PAR + 32 * 64 * 16 * 4;  // ~33.95 MB
    if (ws_size < WS_NEED) return;  // not enough scratch for the fast path

    char* ws = (char*)d_ws;
    unsigned short* Vhi = (unsigned short*)(ws + OFF_VHI);
    unsigned short* Vlo = (unsigned short*)(ws + OFF_VLO);
    unsigned short* Qhi = (unsigned short*)(ws + OFF_QHI);
    unsigned short* Qlo = (unsigned short*)(ws + OFF_QLO);
    float* pos     = (float*)(ws + OFF_POS);
    float* partial = (float*)(ws + OFF_PAR);

    // 1) f32 -> bf16 hi/lo splits
    cvt_split_kernel<<<(32768 + 255) / 256, 256, 0, stream>>>(
        V, (unsigned int*)Vhi, (unsigned int*)Vlo, 32768);
    cvt_split_kernel<<<(4194304 + 255) / 256, 256, 0, stream>>>(
        queue, (unsigned int*)Qhi, (unsigned int*)Qlo, 4194304);

    // 2) positives in full f32
    pos_kernel<<<2, 256, 0, stream>>>(V, L, pos);

    // 3) fused WMMA GEMM + exp-reduction over the 65536-entry queue
    dim3 grid(32, 64);
    gemm_lse_kernel<<<grid, 256, 0, stream>>>(Vhi, Vlo, Qhi, Qlo, partial);

    // 4) assemble logsumexp and the scalar loss
    finalize_kernel<<<1, 512, 0, stream>>>(pos, partial, out);
}
